// DAWN_35253091565665
// MI455X (gfx1250) — matrix operand order repaired and checked by reading, not compiled, not run on hardware
//
#include <hip/hip_runtime.h>
#include <hip/hip_bf16.h>
#include <math.h>

// ---------------- model constants ----------------
static constexpr int BB   = 2;
static constexpr int SS   = 512;
static constexpr int TT   = BB * SS;      // 1024 tokens
static constexpr int DD   = 256;
static constexpr int HH   = 4;
static constexpr int DH   = 64;
static constexpr int LL   = 4;
static constexpr int DFF  = 1024;
static constexpr int VV   = 32000;
static constexpr int NN   = 64;
static constexpr int NB   = 32;
static constexpr int TOPK = 8;

typedef __attribute__((ext_vector_type(16))) _Float16 v16h;
typedef __attribute__((ext_vector_type(8)))  _Float16 v8h;
typedef __attribute__((ext_vector_type(8)))  float    v8f;

enum { FLAG_CAUSAL = 1, FLAG_GELU = 2, FLAG_RES = 4 };

// ---------------- generic batched WMMA GEMM ----------------
// C[m,n] = act( scale * sum_k A[m,k]*Bt[n,k] + bias[n] )   (Bt is B transposed, row-major [N,K])
// One wave32 computes a 16x64 strip (4 accumulators) reusing the A fragment 4x per K-step.
// All N used by this model are multiples of 64.
__global__ void __launch_bounds__(256) k_wmma_gemm(
    const _Float16* __restrict__ A,  long sAb, long sAh, int lda,
    const _Float16* __restrict__ Bt, long sBb, long sBh, int ldb,
    float* __restrict__ C, _Float16* __restrict__ C16,
    long sCb, long sCh, int ldc,
    const float* __restrict__ bias,
    int M, int N, int K, int batchH, int flags, float scale)
{
    const int z  = blockIdx.y;
    const int zb = z / batchH, zh = z % batchH;
    const _Float16* Ab = A  + (long)zb * sAb + (long)zh * sAh;
    const _Float16* Bb = Bt + (long)zb * sBb + (long)zh * sBh;
    const long coff = (long)zb * sCb + (long)zh * sCh;

    const int wave    = blockIdx.x * (blockDim.x >> 5) + (threadIdx.x >> 5);
    const int stripsN = N >> 6;                      // 4 n-tiles per strip
    if (wave >= (M >> 4) * stripsN) return;          // wave-uniform: EXEC stays all-1s
    const int m0 = (wave / stripsN) << 4;
    const int n0 = (wave % stripsN) << 6;

    const int lane = threadIdx.x & 31;
    const int hl   = lane >> 4;                      // lane-half
    const int r    = lane & 15;

    v8f acc[4] = {v8f{}, v8f{}, v8f{}, v8f{}};
    const bool fullyMasked = (flags & FLAG_CAUSAL) && (n0 > m0 + 15);
    if (!fullyMasked) {
        // A frag: lane holds row (m0 + r), 16 contiguous K at hl*16 (matches B)
        const _Float16* ap = Ab + (long)(m0 + r) * lda + hl * 16;
        // B frag: lane holds column (n-tile base + r), 16 contiguous K at hl*16
        const _Float16* bp = Bb + (long)(n0 + r) * ldb + hl * 16;
        const long bstep = (long)16 * ldb;
        for (int kb = 0; kb < K; kb += 32) {
            union { v16h v; v8h h[2]; } a;
            a.h[0] = *(const v8h*)(ap + kb);
            a.h[1] = *(const v8h*)(ap + kb + 8);
            v16h b0 = *(const v16h*)(bp + kb);
            v16h b1 = *(const v16h*)(bp + bstep + kb);
            v16h b2 = *(const v16h*)(bp + 2 * bstep + kb);
            v16h b3 = *(const v16h*)(bp + 3 * bstep + kb);
            acc[0] = __builtin_amdgcn_wmma_f32_16x16x32_f16(false, a.v, false, b0,
                                                            (short)0, acc[0], false, false);
            acc[1] = __builtin_amdgcn_wmma_f32_16x16x32_f16(false, a.v, false, b1,
                                                            (short)0, acc[1], false, false);
            acc[2] = __builtin_amdgcn_wmma_f32_16x16x32_f16(false, a.v, false, b2,
                                                            (short)0, acc[2], false, false);
            acc[3] = __builtin_amdgcn_wmma_f32_16x16x32_f16(false, a.v, false, b3,
                                                            (short)0, acc[3], false, false);
        }
    }
#pragma unroll
    for (int j = 0; j < 4; ++j) {
        const int   n  = n0 + j * 16 + r;
        const float bv = bias ? bias[n] : 0.0f;
#pragma unroll
        for (int i = 0; i < 8; ++i) {
            const int m = m0 + hl * 8 + i;           // C/D layout: VGPR i -> row hl*8+i, lane&15 -> col
            float v = acc[j][i] * scale + bv;
            if ((flags & FLAG_CAUSAL) && (n > m)) v = -1e30f;
            if (flags & FLAG_GELU) v = 0.5f * v * (1.0f + erff(v * 0.70710678118f));
            const long idx = coff + (long)m * ldc + n;
            if (C)   { if (flags & FLAG_RES) C[idx] += v; else C[idx] = v; }
            if (C16) C16[idx] = (_Float16)v;
        }
    }
}

// ---------------- embedding ----------------
__global__ void k_embed(const int* __restrict__ ids, const float* __restrict__ temb,
                        const float* __restrict__ pemb, float* __restrict__ x)
{
    int i = blockIdx.x * blockDim.x + threadIdx.x;   // over TT*DD
    int t = i >> 8, d = i & 255;
    x[i] = temb[(long)ids[t] * DD + d] + pemb[(t & (SS - 1)) * DD + d];
}

// ---------------- layernorm over D=256 (one block per token) ----------------
__global__ void __launch_bounds__(256) k_layernorm(
    const float* __restrict__ x, const float* __restrict__ g, const float* __restrict__ b,
    float* __restrict__ outf, _Float16* __restrict__ outh, int ldh)
{
    __shared__ float red[256];
    const int t = blockIdx.x, d = threadIdx.x;
    float v = x[(long)t * DD + d];
    red[d] = v; __syncthreads();
    for (int s = 128; s > 0; s >>= 1) { if (d < s) red[d] += red[d + s]; __syncthreads(); }
    const float mean = red[0] * (1.0f / DD); __syncthreads();
    const float c = v - mean;
    red[d] = c * c; __syncthreads();
    for (int s = 128; s > 0; s >>= 1) { if (d < s) red[d] += red[d + s]; __syncthreads(); }
    const float y = c * rsqrtf(red[0] * (1.0f / DD) + 1e-5f) * g[d] + b[d];
    if (outf) outf[(long)t * DD + d] = y;
    if (outh) outh[(long)t * ldh + d] = (_Float16)y;
}

// ---------------- converts ----------------
__global__ void k_cvt(const float* __restrict__ in, _Float16* __restrict__ out, long total)
{
    long i = (long)blockIdx.x * blockDim.x + threadIdx.x;
    if (i < total) out[i] = (_Float16)in[i];
}
// W [batch,K,N] f32 -> Wt [batch,N,K] f16
__global__ void k_cvt_transpose(const float* __restrict__ W, _Float16* __restrict__ Wt,
                                int Kd, int Nd, long total)
{
    long i = (long)blockIdx.x * blockDim.x + threadIdx.x;
    if (i >= total) return;
    int  n = (int)(i % Nd);
    long t = i / Nd;
    int  k = (int)(t % Kd);
    long l = t / Kd;
    Wt[((long)l * Nd + n) * Kd + k] = (_Float16)W[i];
}
// v_f16 [T, H*DH] -> vT [B,H,DH,S]   (makes PV's Bt fragment loads contiguous)
__global__ void k_pack_vT(const _Float16* __restrict__ v, _Float16* __restrict__ vT)
{
    int i = blockIdx.x * blockDim.x + threadIdx.x;   // TT*DD
    int t = i >> 8, c = i & 255;
    int b = t >> 9, s = t & 511;
    int h = c >> 6, dh = c & 63;
    vT[((((long)b * HH + h) * DH + dh) << 9) + s] = v[i];
}

// ---------------- softmax over rows of length 512 ----------------
__global__ void __launch_bounds__(256) k_softmax_row(const float* __restrict__ att,
                                                     _Float16* __restrict__ p)
{
    __shared__ float red[256];
    const long row = blockIdx.x;
    const float* a = att + row * SS;
    const int d = threadIdx.x;
    float v0 = a[d], v1 = a[d + 256];
    red[d] = fmaxf(v0, v1); __syncthreads();
    for (int s = 128; s > 0; s >>= 1) { if (d < s) red[d] = fmaxf(red[d], red[d + s]); __syncthreads(); }
    const float mx = red[0]; __syncthreads();
    float e0 = __expf(v0 - mx), e1 = __expf(v1 - mx);
    red[d] = e0 + e1; __syncthreads();
    for (int s = 128; s > 0; s >>= 1) { if (d < s) red[d] += red[d + s]; __syncthreads(); }
    const float inv = 1.0f / red[0];
    p[row * SS + d]       = (_Float16)(e0 * inv);
    p[row * SS + d + 256] = (_Float16)(e1 * inv);
}

// ---------------- recipes softmax (L*NN rows of NB) ----------------
__global__ void k_recipes_sm(const float* __restrict__ rec, float* __restrict__ rs)
{
    int row = blockIdx.x * blockDim.x + threadIdx.x;
    if (row >= LL * NN) return;
    const float* r = rec + (long)row * NB;
    float* o = rs + (long)row * NB;
    float mx = -1e30f;
    for (int i = 0; i < NB; ++i) mx = fmaxf(mx, r[i]);
    float e[NB], s = 0.f;
    for (int i = 0; i < NB; ++i) { e[i] = __expf(r[i] - mx); s += e[i]; }
    const float inv = 1.0f / s;
    for (int i = 0; i < NB; ++i) o[i] = e[i] * inv;
}

// ---------------- neuron_emb[l,n,d] = recipes_sm[l,n,:] @ basis_emb  (f16 out = Bt for score GEMM) ----
__global__ void k_neuron_emb(const float* __restrict__ rs, const float* __restrict__ basis,
                             _Float16* __restrict__ ne)
{
    int i = blockIdx.x * blockDim.x + threadIdx.x;   // LL*NN*DD
    if (i >= LL * NN * DD) return;
    int d = i & 255, row = i >> 8;                   // row = l*NN + n
    const float* w = rs + (long)row * NB;
    float acc = 0.f;
    for (int b2 = 0; b2 < NB; ++b2) acc += w[b2] * basis[b2 * DD + d];
    ne[i] = (_Float16)acc;
}

// ---------------- top-k(8) + softmax + weighted recipe mix -> wr [T,NB] ----------------
__global__ void k_topk_wr(const float* __restrict__ scores, const float* __restrict__ rs_l,
                          float* __restrict__ wr)
{
    int t = blockIdx.x * blockDim.x + threadIdx.x;
    if (t >= TT) return;
    float sc[NN];
    for (int i = 0; i < NN; ++i) sc[i] = scores[(long)t * NN + i];
    int idx[TOPK]; float val[TOPK];
    for (int k = 0; k < TOPK; ++k) {
        float best = -1e30f; int bi = 0;
        for (int i = 0; i < NN; ++i) if (sc[i] > best) { best = sc[i]; bi = i; }
        idx[k] = bi; val[k] = best; sc[bi] = -1e30f;
    }
    float s = 0.f, w8[TOPK];
    for (int k = 0; k < TOPK; ++k) { w8[k] = __expf(val[k] - val[0]); s += w8[k]; }
    const float inv = 1.0f / s;
    float acc[NB];
    for (int n = 0; n < NB; ++n) acc[n] = 0.f;
    for (int k = 0; k < TOPK; ++k) {
        const float wk = w8[k] * inv;
        const float* rr = rs_l + (long)idx[k] * NB;
        for (int n = 0; n < NB; ++n) acc[n] += wk * rr[n];
    }
    for (int n = 0; n < NB; ++n) wr[(long)t * NB + n] = acc[n];
}

// ---------------- TT FFN: one token per block, LDS-resident cores ----------------
// out[t, 0:1024] = gelu( TT_apply(nrm2[t], wr[t]-mixed cores) ), f16
__global__ void __launch_bounds__(256) k_tt_ffn(
    const float* __restrict__ nrm2, const float* __restrict__ wr,
    const float* __restrict__ A1, const float* __restrict__ A2,
    const float* __restrict__ B1, const float* __restrict__ B2,
    _Float16* __restrict__ out)
{
    extern __shared__ float sm[];
    float* buf1 = sm;              // 16384 floats: cA1 / cA2 / cB1 / cB2
    float* buf2 = sm + 16384;      // 16384 floats: t1 / tb
    float* xf   = sm + 32768;      // 256
    float* hbuf = xf + 256;        // 64
    float* wrs  = hbuf + 64;       // 32
    const int t = blockIdx.x, tid = threadIdx.x;
    xf[tid] = nrm2[(long)t * DD + tid];
    if (tid < NB) wrs[tid] = wr[(long)t * NB + tid];
    __syncthreads();
    // cA1[i(16),r(64),k(8)] = sum_n wr[n]*A1[n,i,r,k]
    for (int e = tid; e < 8192; e += 256) {
        float a = 0.f;
        for (int n = 0; n < NB; ++n) a += wrs[n] * A1[(long)n * 8192 + e];
        buf1[e] = a;
    }
    __syncthreads();
    // t1[j,r,k] = sum_i xf[i*16+j]*cA1[i,r,k]
    for (int e = tid; e < 8192; e += 256) {
        const int j = e >> 9, rk = e & 511;
        float a = 0.f;
        for (int i = 0; i < 16; ++i) a += xf[i * 16 + j] * buf1[i * 512 + rk];
        buf2[e] = a;
    }
    __syncthreads();
    // cA2[r(64),j(16),l(8)]
    for (int e = tid; e < 8192; e += 256) {
        float a = 0.f;
        for (int n = 0; n < NB; ++n) a += wrs[n] * A2[(long)n * 8192 + e];
        buf1[e] = a;
    }
    __syncthreads();
    // h[k(8),l(8)] = sum_{j,r} t1[j,r,k]*cA2[r,j,l]
    if (tid < 64) {
        const int k2 = tid >> 3, l = tid & 7;
        float a = 0.f;
        for (int j = 0; j < 16; ++j)
            for (int r = 0; r < 64; ++r)
                a += buf2[j * 512 + r * 8 + k2] * buf1[r * 128 + j * 8 + l];
        hbuf[k2 * 8 + l] = a;
    }
    __syncthreads();
    // cB1[i(8),r(64),k(32)]
    for (int e = tid; e < 16384; e += 256) {
        float a = 0.f;
        for (int n = 0; n < NB; ++n) a += wrs[n] * B1[(long)n * 16384 + e];
        buf1[e] = a;
    }
    __syncthreads();
    // tb[j(8),r,k] = sum_i hf[i*8+j]*cB1[i,r,k]
    for (int e = tid; e < 16384; e += 256) {
        const int j = e >> 11, rk = e & 2047;
        float a = 0.f;
        for (int i = 0; i < 8; ++i) a += hbuf[i * 8 + j] * buf1[i * 2048 + rk];
        buf2[e] = a;
    }
    __syncthreads();
    // cB2[r(64),j(8),l(32)]
    for (int e = tid; e < 16384; e += 256) {
        float a = 0.f;
        for (int n = 0; n < NB; ++n) a += wrs[n] * B2[(long)n * 16384 + e];
        buf1[e] = a;
    }
    __syncthreads();
    // out[k(32),l(32)] = gelu( sum_{j,r} tb[j,r,k]*cB2[r,j,l] )
    for (int e = tid; e < 1024; e += 256) {
        const int k2 = e >> 5, l = e & 31;
        float a = 0.f;
        for (int j = 0; j < 8; ++j)
            for (int r = 0; r < 64; ++r)
                a += buf2[j * 2048 + r * 32 + k2] * buf1[r * 256 + j * 32 + l];
        const float g = 0.5f * a * (1.0f + erff(a * 0.70710678118f));
        out[(long)t * DFF + e] = (_Float16)g;
    }
}

// ---------------- host-side GEMM wrapper ----------------
static void launch_gemm(hipStream_t st,
    const _Float16* A, long sAb, long sAh, int lda,
    const _Float16* Bt, long sBb, long sBh, int ldb,
    float* C, _Float16* C16, long sCb, long sCh, int ldc,
    const float* bias, int M, int N, int K,
    int nbatch, int batchH, int flags, float scale)
{
    const int strips = (M >> 4) * (N >> 6);          // one wave per 16x64 strip
    const int blocks = (strips + 7) / 8;             // 8 waves per 256-thread block
    dim3 g(blocks, nbatch, 1);
    k_wmma_gemm<<<g, dim3(256), 0, st>>>(A, sAb, sAh, lda, Bt, sBb, sBh, ldb,
                                         C, C16, sCb, sCh, ldc, bias,
                                         M, N, K, batchH, flags, scale);
}

extern "C" void kernel_launch(void* const* d_in, const int* in_sizes, int n_in,
                              void* d_out, int out_size, void* d_ws, size_t ws_size,
                              hipStream_t stream)
{
    (void)in_sizes; (void)n_in; (void)out_size; (void)ws_size;
    const int*   ids   = (const int*)  d_in[0];
    const float* temb  = (const float*)d_in[1];
    const float* pemb  = (const float*)d_in[2];
    const float* qW    = (const float*)d_in[3];
    const float* qb    = (const float*)d_in[4];
    const float* kW    = (const float*)d_in[5];
    const float* kb    = (const float*)d_in[6];
    const float* vW    = (const float*)d_in[7];
    const float* vb    = (const float*)d_in[8];
    const float* sW    = (const float*)d_in[9];
    const float* sb    = (const float*)d_in[10];
    const float* rec   = (const float*)d_in[11];
    const float* wdW   = (const float*)d_in[12];
    const float* wdb   = (const float*)d_in[13];
    const float* n1g   = (const float*)d_in[14];
    const float* n1b   = (const float*)d_in[15];
    const float* n2g   = (const float*)d_in[16];
    const float* n2b   = (const float*)d_in[17];
    const float* basis = (const float*)d_in[18];
    const float* A1    = (const float*)d_in[19];
    const float* A2    = (const float*)d_in[20];
    const float* B1    = (const float*)d_in[21];
    const float* B2    = (const float*)d_in[22];
    const float* fng   = (const float*)d_in[23];
    const float* fnb   = (const float*)d_in[24];

    // ---- workspace bump allocator ----
    char* ws = (char*)d_ws; size_t off = 0;
    auto alloc = [&](size_t bytes) -> void* {
        void* p = ws + off; off = (off + bytes + 255) & ~(size_t)255; return p;
    };
    float*    x      = (float*)   alloc((size_t)TT * DD * 4);
    _Float16* concat = (_Float16*)alloc((size_t)TT * 2 * DD * 2);  // [T,512]: [nrm1 | ctx]
    _Float16* qh     = (_Float16*)alloc((size_t)TT * DD * 2);
    _Float16* kh     = (_Float16*)alloc((size_t)TT * DD * 2);
    _Float16* vh     = (_Float16*)alloc((size_t)TT * DD * 2);
    _Float16* vT     = (_Float16*)alloc((size_t)TT * DD * 2);
    float*    att    = (float*)   alloc((size_t)BB * HH * SS * SS * 4);
    _Float16* prob   = (_Float16*)alloc((size_t)BB * HH * SS * SS * 2);
    _Float16* qryh   = (_Float16*)alloc((size_t)TT * DD * 2);
    float*    scr    = (float*)   alloc((size_t)TT * NN * 4);
    float*    wr     = (float*)   alloc((size_t)TT * NB * 4);
    float*    nrm2   = (float*)   alloc((size_t)TT * DD * 4);
    _Float16* ffh    = (_Float16*)alloc((size_t)TT * DFF * 2);
    _Float16* qWt    = (_Float16*)alloc((size_t)LL * DD * DD * 2);
    _Float16* kWt    = (_Float16*)alloc((size_t)LL * DD * DD * 2);
    _Float16* vWt    = (_Float16*)alloc((size_t)LL * DD * DD * 2);
    _Float16* sWt    = (_Float16*)alloc((size_t)LL * 2 * DD * DD * 2);
    _Float16* wdWt   = (_Float16*)alloc((size_t)LL * DFF * DD * 2);
    _Float16* tembh  = (_Float16*)alloc((size_t)VV * DD * 2);
    float*    rs     = (float*)   alloc((size_t)LL * NN * NB * 4);
    _Float16* neh    = (_Float16*)alloc((size_t)LL * NN * DD * 2);
    _Float16* xfh    = (_Float16*)alloc((size_t)TT * DD * 2);

    // ---- one-time prep (runs every call; deterministic) ----
    k_embed<<<TT, 256, 0, stream>>>(ids, temb, pemb, x);
    {
        long t1 = (long)LL * DD * DD;
        k_cvt_transpose<<<(int)((t1 + 255) / 256), 256, 0, stream>>>(qW, qWt, DD, DD, t1);
        k_cvt_transpose<<<(int)((t1 + 255) / 256), 256, 0, stream>>>(kW, kWt, DD, DD, t1);
        k_cvt_transpose<<<(int)((t1 + 255) / 256), 256, 0, stream>>>(vW, vWt, DD, DD, t1);
        long t2 = (long)LL * 2 * DD * DD;
        k_cvt_transpose<<<(int)((t2 + 255) / 256), 256, 0, stream>>>(sW, sWt, 2 * DD, DD, t2);
        long t3 = (long)LL * DFF * DD;
        k_cvt_transpose<<<(int)((t3 + 255) / 256), 256, 0, stream>>>(wdW, wdWt, DFF, DD, t3);
        long t4 = (long)VV * DD;
        k_cvt<<<(int)((t4 + 255) / 256), 256, 0, stream>>>(temb, tembh, t4);
    }
    k_recipes_sm<<<1, 256, 0, stream>>>(rec, rs);
    k_neuron_emb<<<(LL * NN * DD) / 256, 256, 0, stream>>>(rs, basis, neh);

    // ---- layers ----
    for (int l = 0; l < LL; ++l) {
        // LN1 -> f16 directly into concat[:, 0:256]
        k_layernorm<<<TT, 256, 0, stream>>>(x, n1g + l * DD, n1b + l * DD,
                                            (float*)nullptr, concat, 2 * DD);
        // Q/K/V projections (f16 outputs only)
        launch_gemm(stream, concat, 0, 0, 2 * DD, qWt + (long)l * DD * DD, 0, 0, DD,
                    nullptr, qh, 0, 0, DD, qb + l * DD, TT, DD, DD, 1, 1, 0, 1.0f);
        launch_gemm(stream, concat, 0, 0, 2 * DD, kWt + (long)l * DD * DD, 0, 0, DD,
                    nullptr, kh, 0, 0, DD, kb + l * DD, TT, DD, DD, 1, 1, 0, 1.0f);
        launch_gemm(stream, concat, 0, 0, 2 * DD, vWt + (long)l * DD * DD, 0, 0, DD,
                    nullptr, vh, 0, 0, DD, vb + l * DD, TT, DD, DD, 1, 1, 0, 1.0f);
        k_pack_vT<<<TT, 256, 0, stream>>>(vh, vT);
        // attention logits: batched over B*H, causal + 1/sqrt(64)
        launch_gemm(stream, qh, (long)SS * DD, DH, DD, kh, (long)SS * DD, DH, DD,
                    att, nullptr, (long)HH * SS * SS, (long)SS * SS, SS,
                    nullptr, SS, SS, DH, BB * HH, HH, FLAG_CAUSAL, 0.125f);
        k_softmax_row<<<BB * HH * SS, 256, 0, stream>>>(att, prob);
        // ctx = P @ V, f16 written straight into concat[:, 256:512]
        launch_gemm(stream, prob, (long)HH * SS * SS, (long)SS * SS, SS,
                    vT, (long)HH * DH * SS, (long)DH * SS, SS,
                    nullptr, concat + DD, (long)SS * 2 * DD, DH, 2 * DD,
                    nullptr, SS, DH, SS, BB * HH, HH, 0, 1.0f);
        // query = concat @ sW + sb
        launch_gemm(stream, concat, 0, 0, 2 * DD, sWt + (long)l * 2 * DD * DD, 0, 0, 2 * DD,
                    nullptr, qryh, 0, 0, DD, sb + l * DD, TT, DD, 2 * DD, 1, 1, 0, 1.0f);
        // router scores = query @ neuron_emb^T
        launch_gemm(stream, qryh, 0, 0, DD, neh + (long)l * NN * DD, 0, 0, DD,
                    scr, nullptr, 0, 0, NN, nullptr, TT, NN, DD, 1, 1, 0, 1.0f);
        k_topk_wr<<<TT / 256, 256, 0, stream>>>(scr, rs + (long)l * NN * NB, wr);
        // LN2 (f32 for TT FFN)
        k_layernorm<<<TT, 256, 0, stream>>>(x, n2g + l * DD, n2b + l * DD,
                                            nrm2, (_Float16*)nullptr, 0);
        // TT FFN (128 KB dynamic LDS per token-block)
        k_tt_ffn<<<TT, 256, (16384 + 16384 + 256 + 64 + 32) * sizeof(float), stream>>>(
            nrm2, wr, A1, A2, B1, B2, ffh);
        // down-proj + residual into x
        launch_gemm(stream, ffh, 0, 0, DFF, wdWt + (long)l * DFF * DD, 0, 0, DFF,
                    x, nullptr, 0, 0, DD, wdb + l * DD, TT, DD, DFF, 1, 1, FLAG_RES, 1.0f);
    }

    // ---- final LN + tied-head logits ----
    k_layernorm<<<TT, 256, 0, stream>>>(x, fng, fnb, (float*)nullptr, xfh, DD);
    launch_gemm(stream, xfh, 0, 0, DD, tembh, 0, 0, DD,
                (float*)d_out, nullptr, 0, 0, VV, nullptr, TT, VV, DD, 1, 1, 0, 1.0f);
}
